// FactorMPNN_77558519431291
// MI455X (gfx1250) — matrix operand order repaired and checked by reading, not compiled, not run on hardware
//
#include <hip/hip_runtime.h>

typedef __attribute__((ext_vector_type(16))) __bf16 v16bf;
typedef __attribute__((ext_vector_type(8)))  float  v8f;

union V16 { v16bf v; __bf16 h[16]; uint4 q[2]; };

#define HID 64

static __device__ __forceinline__ void lds_fence() {
  asm volatile("s_wait_dscnt 0" ::: "memory");
}

static __device__ __forceinline__ void async_fence() {
  asm volatile("s_wait_asynccnt 0" ::: "memory");
}

// gfx1250 async copy: global -> LDS, 16 bytes, tracked by ASYNCcnt.
// lds_off = LDS byte address (low 32 bits of the flat shared pointer).
static __device__ __forceinline__ void async_copy_b128(unsigned lds_off, const void* gptr) {
  asm volatile("global_load_async_to_lds_b128 %0, %1, off"
               :: "v"(lds_off), "v"(gptr) : "memory");
}

static __device__ __forceinline__ void wmma_bf16(v16bf a, v16bf b, v8f& c) {
  c = __builtin_amdgcn_wmma_f32_16x16x32_bf16(false, a, false, b, (short)0, c, false, false);
}

// B fragment (32x16 K-major weights) from LDS, weights stored transposed [64][K]:
// lane l: col n = 16*nt + (l&15), K runs of 8 at kt*32 + 8*(l>>4) and kt*32 + 16 + 8*(l>>4)
static __device__ __forceinline__ v16bf load_b_frag(const __bf16* Wt, int K, int nt, int kt,
                                                    int r, int hh) {
  const uint4* p = (const uint4*)(Wt + (nt * 16 + r) * K + kt * 32 + hh * 8);
  V16 t;
  t.q[0] = p[0];
  t.q[1] = p[2];
  return t.v;
}

static __device__ __forceinline__ void cvt8(__bf16* dst, float4 a, float4 b) {
  dst[0] = (__bf16)a.x; dst[1] = (__bf16)a.y; dst[2] = (__bf16)a.z; dst[3] = (__bf16)a.w;
  dst[4] = (__bf16)b.x; dst[5] = (__bf16)b.y; dst[6] = (__bf16)b.z; dst[7] = (__bf16)b.w;
}

static __device__ __forceinline__ void cvt8s(__bf16* dst, float4 a, float4 b, float s) {
  dst[0] = (__bf16)(a.x * s); dst[1] = (__bf16)(a.y * s);
  dst[2] = (__bf16)(a.z * s); dst[3] = (__bf16)(a.w * s);
  dst[4] = (__bf16)(b.x * s); dst[5] = (__bf16)(b.y * s);
  dst[6] = (__bf16)(b.z * s); dst[7] = (__bf16)(b.w * s);
}

// X row = [node_attr[s] (64) | node_attr[d] (64) | edge_attr[e] (64)]; c is a multiple of 8
static __device__ __forceinline__ const float* secptr(const float* __restrict__ na,
                                                      const float* __restrict__ ea,
                                                      int s, int d, long e, int c) {
  if (c < 64)  return na + (long)s * HID + c;
  if (c < 128) return na + (long)d * HID + (c - 64);
  return ea + e * HID + (c - 128);
}

__global__ void zero_ws_kernel(float4* __restrict__ p, long n4) {
  long t = (long)blockIdx.x * blockDim.x + threadIdx.x;
  long stride = (long)gridDim.x * blockDim.x;
  float4 z; z.x = 0.f; z.y = 0.f; z.z = 0.f; z.w = 0.f;
  for (; t < n4; t += stride) p[t] = z;
}

// Wt[n*K + k] = bf16(W[k*Nc + n])
__global__ void transpose_w_kernel(const float* __restrict__ W, __bf16* __restrict__ Wt,
                                   int K, int Nc) {
  int total = K * Nc;
  for (int t = blockIdx.x * blockDim.x + threadIdx.x; t < total; t += gridDim.x * blockDim.x) {
    int n = t / K;
    int k = t - n * K;
    Wt[t] = (__bf16)W[(long)k * Nc + n];
  }
}

__global__ __launch_bounds__(256) void edge_flow_kernel(
    const float* __restrict__ node_attr, const float* __restrict__ edge_attr,
    const int* __restrict__ eidx, long E, int swp, int bidir,
    const __bf16* __restrict__ W1t, const float* __restrict__ b1,
    const __bf16* __restrict__ W2t, const float* __restrict__ b2,
    float* __restrict__ sum, float* __restrict__ cnt) {
  __shared__ __bf16 sW1[192 * 64];
  __shared__ __bf16 sW2[64 * 64];
  __shared__ __bf16 sH[8][16 * 72];

  {
    unsigned o1 = (unsigned)(size_t)(void*)sW1;
    for (int i = threadIdx.x; i < 192 * 64 / 8; i += blockDim.x)
      async_copy_b128(o1 + i * 16, (const char*)W1t + i * 16);
    unsigned o2 = (unsigned)(size_t)(void*)sW2;
    for (int i = threadIdx.x; i < 64 * 64 / 8; i += blockDim.x)
      async_copy_b128(o2 + i * 16, (const char*)W2t + i * 16);
    async_fence();
  }
  __syncthreads();

  const int lane = threadIdx.x & 31;
  const int wv = threadIdx.x >> 5;
  const int r = lane & 15;
  const int hh = lane >> 4;
  __bf16* hT = &sH[wv][0];

  // tile-invariant biases
  float bias1[4], bias2[4];
#pragma unroll
  for (int nt = 0; nt < 4; ++nt) {
    bias1[nt] = b1[nt * 16 + r];
    bias2[nt] = b2[nt * 16 + r];
  }

  const long ntiles = E >> 4;
  const long stride = (long)gridDim.x * 8;

  for (long tile = (long)blockIdx.x * 8 + wv; tile < ntiles; tile += stride) {
    const long base = tile << 4;
    const long er = base + r;
    const int i0 = eidx[er], i1 = eidx[E + er];
    const int sr = swp ? i1 : i0;
    const int dr = swp ? i0 : i1;
    int sj[8], dj[8];
#pragma unroll
    for (int j = 0; j < 8; ++j) {
      long m = base + hh * 8 + j;
      int a0 = eidx[m], a1 = eidx[E + m];
      sj[j] = swp ? a1 : a0;
      dj[j] = swp ? a0 : a1;
    }

    // ---- layer 1: [16 x 192] @ [192 x 64] + b1 ----
    v8f acc[4];
#pragma unroll
    for (int nt = 0; nt < 4; ++nt)
#pragma unroll
      for (int j = 0; j < 8; ++j) acc[nt][j] = bias1[nt];
#pragma unroll
    for (int kt = 0; kt < 6; ++kt) {
      int c0 = kt * 32 + hh * 8;
      const float* p0 = secptr(node_attr, edge_attr, sr, dr, er, c0);
      const float* p1 = secptr(node_attr, edge_attr, sr, dr, er, c0 + 16);
      float4 u0 = *(const float4*)p0;
      float4 u1 = *(const float4*)(p0 + 4);
      float4 w0 = *(const float4*)p1;
      float4 w1 = *(const float4*)(p1 + 4);
      V16 A;
      cvt8(&A.h[0], u0, u1);
      cvt8(&A.h[8], w0, w1);
#pragma unroll
      for (int nt = 0; nt < 4; ++nt) {
        v16bf B = load_b_frag(sW1, 192, nt, kt, r, hh);
        wmma_bf16(A.v, B, acc[nt]);
      }
    }

    // ---- relu -> LDS (row-major 16x64, stride 72) ----
    lds_fence();
#pragma unroll
    for (int nt = 0; nt < 4; ++nt)
#pragma unroll
      for (int j = 0; j < 8; ++j)
        hT[(hh * 8 + j) * 72 + nt * 16 + r] = (__bf16)fmaxf(acc[nt][j], 0.0f);
    lds_fence();

    // ---- layer 2: [16 x 64] @ [64 x 64] + b2 ----
    v8f acc2[4];
#pragma unroll
    for (int nt = 0; nt < 4; ++nt)
#pragma unroll
      for (int j = 0; j < 8; ++j) acc2[nt][j] = bias2[nt];
#pragma unroll
    for (int kt = 0; kt < 2; ++kt) {
      int c0 = kt * 32 + hh * 8;
      V16 A;
      A.q[0] = *(const uint4*)(hT + r * 72 + c0);
      A.q[1] = *(const uint4*)(hT + r * 72 + c0 + 16);
#pragma unroll
      for (int nt = 0; nt < 4; ++nt) {
        v16bf B = load_b_frag(sW2, 64, nt, kt, r, hh);
        wmma_bf16(A.v, B, acc2[nt]);
      }
    }

    // ---- scatter-add (mean numerator) + counts ----
#pragma unroll
    for (int nt = 0; nt < 4; ++nt)
#pragma unroll
      for (int j = 0; j < 8; ++j) {
        float v = acc2[nt][j];
        atomicAdd(sum + (long)dj[j] * HID + nt * 16 + r, v);
        if (bidir) atomicAdd(sum + (long)sj[j] * HID + nt * 16 + r, v);
      }
    if (r == 0) {
#pragma unroll
      for (int j = 0; j < 8; ++j) {
        atomicAdd(cnt + dj[j], 1.0f);
        if (bidir) atomicAdd(cnt + sj[j], 1.0f);
      }
    }
  }
}

__global__ __launch_bounds__(256) void node_mlp_kernel(
    const float* __restrict__ sums, const float* __restrict__ cnts,
    const __bf16* __restrict__ TW1t, const float* __restrict__ tb1,
    const __bf16* __restrict__ TW2t, const float* __restrict__ tb2,
    float* __restrict__ out, long Nn) {
  __shared__ __bf16 sW1[256 * 64];
  __shared__ __bf16 sW2[64 * 64];
  __shared__ __bf16 sH[8][16 * 72];

  {
    unsigned o1 = (unsigned)(size_t)(void*)sW1;
    for (int i = threadIdx.x; i < 256 * 64 / 8; i += blockDim.x)
      async_copy_b128(o1 + i * 16, (const char*)TW1t + i * 16);
    unsigned o2 = (unsigned)(size_t)(void*)sW2;
    for (int i = threadIdx.x; i < 64 * 64 / 8; i += blockDim.x)
      async_copy_b128(o2 + i * 16, (const char*)TW2t + i * 16);
    async_fence();
  }
  __syncthreads();

  const int lane = threadIdx.x & 31;
  const int wv = threadIdx.x >> 5;
  const int r = lane & 15;
  const int hh = lane >> 4;
  __bf16* hT = &sH[wv][0];

  float bias1[4], bias2[4];
#pragma unroll
  for (int nt = 0; nt < 4; ++nt) {
    bias1[nt] = tb1[nt * 16 + r];
    bias2[nt] = tb2[nt * 16 + r];
  }

  const long ntiles = Nn >> 4;
  const long stride = (long)gridDim.x * 8;

  for (long tile = (long)blockIdx.x * 8 + wv; tile < ntiles; tile += stride) {
    const long nbase = tile << 4;
    const long node = nbase + r;

    // ---- layer 1: agg[16 x 256] @ [256 x 64] + tb1 ----
    v8f acc[4];
#pragma unroll
    for (int nt = 0; nt < 4; ++nt)
#pragma unroll
      for (int j = 0; j < 8; ++j) acc[nt][j] = bias1[nt];
#pragma unroll
    for (int kt = 0; kt < 8; ++kt) {
      int c0 = kt * 32 + hh * 8;       // both 8-runs (c0, c0+16) lie in one flow's 64-block
      int f = c0 >> 6;
      int cc = c0 & 63;
      float inv = 1.0f / fmaxf(cnts[(long)f * Nn + node], 1.0f);
      const float* sp = sums + ((long)f * Nn + node) * HID + cc;
      float4 u0 = *(const float4*)sp;
      float4 u1 = *(const float4*)(sp + 4);
      float4 w0 = *(const float4*)(sp + 16);
      float4 w1 = *(const float4*)(sp + 20);
      V16 A;
      cvt8s(&A.h[0], u0, u1, inv);
      cvt8s(&A.h[8], w0, w1, inv);
#pragma unroll
      for (int nt = 0; nt < 4; ++nt) {
        v16bf B = load_b_frag(sW1, 256, nt, kt, r, hh);
        wmma_bf16(A.v, B, acc[nt]);
      }
    }

    lds_fence();
#pragma unroll
    for (int nt = 0; nt < 4; ++nt)
#pragma unroll
      for (int j = 0; j < 8; ++j)
        hT[(hh * 8 + j) * 72 + nt * 16 + r] = (__bf16)fmaxf(acc[nt][j], 0.0f);
    lds_fence();

    // ---- layer 2: [16 x 64] @ [64 x 64] + tb2 ----
    v8f acc2[4];
#pragma unroll
    for (int nt = 0; nt < 4; ++nt)
#pragma unroll
      for (int j = 0; j < 8; ++j) acc2[nt][j] = bias2[nt];
#pragma unroll
    for (int kt = 0; kt < 2; ++kt) {
      int c0 = kt * 32 + hh * 8;
      V16 A;
      A.q[0] = *(const uint4*)(hT + r * 72 + c0);
      A.q[1] = *(const uint4*)(hT + r * 72 + c0 + 16);
#pragma unroll
      for (int nt = 0; nt < 4; ++nt) {
        v16bf B = load_b_frag(sW2, 64, nt, kt, r, hh);
        wmma_bf16(A.v, B, acc2[nt]);
      }
    }

#pragma unroll
    for (int nt = 0; nt < 4; ++nt)
#pragma unroll
      for (int j = 0; j < 8; ++j)
        out[(nbase + hh * 8 + j) * HID + nt * 16 + r] = acc2[nt][j];
  }
}

extern "C" void kernel_launch(void* const* d_in, const int* in_sizes, int n_in,
                              void* d_out, int out_size, void* d_ws, size_t ws_size,
                              hipStream_t stream) {
  const float* node_attr = (const float*)d_in[0];
  const float* ea[3] = {(const float*)d_in[1], (const float*)d_in[2], (const float*)d_in[3]};
  const int* ei[3] = {(const int*)d_in[4], (const int*)d_in[5], (const int*)d_in[6]};
  const long Nn = (long)in_sizes[0] / HID;
  const long E = (long)in_sizes[4] / 2;

  const float* W1[4] = {(const float*)d_in[8], (const float*)d_in[12],
                        (const float*)d_in[16], (const float*)d_in[20]};
  const float* b1[4] = {(const float*)d_in[9], (const float*)d_in[13],
                        (const float*)d_in[17], (const float*)d_in[21]};
  const float* W2[4] = {(const float*)d_in[10], (const float*)d_in[14],
                        (const float*)d_in[18], (const float*)d_in[22]};
  const float* b2[4] = {(const float*)d_in[11], (const float*)d_in[15],
                        (const float*)d_in[19], (const float*)d_in[23]};
  const float* tW1 = (const float*)d_in[24];
  const float* tb1 = (const float*)d_in[25];
  const float* tW2 = (const float*)d_in[26];
  const float* tb2 = (const float*)d_in[27];

  // workspace layout: sums [4][N][64] f32 | counts [4][N] f32 | bf16 transposed weights
  char* ws = (char*)d_ws;
  float* sums = (float*)ws;
  float* cnts = (float*)(ws + (size_t)4 * Nn * HID * sizeof(float));
  __bf16* wb = (__bf16*)(ws + ((size_t)4 * Nn * HID + 4 * Nn) * sizeof(float));
  __bf16 *W1t[4], *W2t[4];
  size_t off = 0;
  for (int t = 0; t < 4; ++t) {
    W1t[t] = wb + off; off += 192 * 64;
    W2t[t] = wb + off; off += 64 * 64;
  }
  __bf16* TW1t = wb + off; off += 256 * 64;
  __bf16* TW2t = wb + off; off += 64 * 64;

  // 1) zero accumulators (sums + counts): 4*Nn*65 floats, multiple of 4
  long n4 = Nn * 65;  // (4*Nn*64 + 4*Nn)/4
  zero_ws_kernel<<<2048, 256, 0, stream>>>((float4*)sums, n4);

  // 2) convert + transpose weights to bf16 [nout][nin]
  for (int t = 0; t < 4; ++t) {
    transpose_w_kernel<<<96, 256, 0, stream>>>(W1[t], W1t[t], 192, 64);
    transpose_w_kernel<<<32, 256, 0, stream>>>(W2[t], W2t[t], 64, 64);
  }
  transpose_w_kernel<<<128, 256, 0, stream>>>(tW1, TW1t, 256, 64);
  transpose_w_kernel<<<32, 256, 0, stream>>>(tW2, TW2t, 64, 64);

  // 3) four edge passes: fwd (t, no swap), bwd (t, swap), soc (s, bidir), view (v, bidir)
  edge_flow_kernel<<<1024, 256, 0, stream>>>(node_attr, ea[0], ei[0], E, 0, 0,
                                             W1t[0], b1[0], W2t[0], b2[0],
                                             sums + (size_t)0 * Nn * HID, cnts + 0 * Nn);
  edge_flow_kernel<<<1024, 256, 0, stream>>>(node_attr, ea[0], ei[0], E, 1, 0,
                                             W1t[1], b1[1], W2t[1], b2[1],
                                             sums + (size_t)1 * Nn * HID, cnts + 1 * Nn);
  edge_flow_kernel<<<1024, 256, 0, stream>>>(node_attr, ea[1], ei[1], E, 0, 1,
                                             W1t[2], b1[2], W2t[2], b2[2],
                                             sums + (size_t)2 * Nn * HID, cnts + 2 * Nn);
  edge_flow_kernel<<<1024, 256, 0, stream>>>(node_attr, ea[2], ei[2], E, 0, 1,
                                             W1t[3], b1[3], W2t[3], b2[3],
                                             sums + (size_t)3 * Nn * HID, cnts + 3 * Nn);

  // 4) node MLP on concatenated means
  node_mlp_kernel<<<512, 256, 0, stream>>>(sums, cnts, TW1t, tb1, TW2t, tb2,
                                           (float*)d_out, Nn);
}
